// GCN_18614388261059
// MI455X (gfx1250) — matrix operand order repaired and checked by reading, not compiled, not run on hardware
//
#include <hip/hip_runtime.h>
#include <stdint.h>

// ---------------------------------------------------------------------------
// GCN forward for MI455X (gfx1250):
//   - bf16 WMMA GEMMs (v_wmma_f32_16x16x32_bf16, f32 accumulate)
//   - GEMM1 stages B in LDS via gfx1250 async-to-LDS copies (double buffered)
//   - per k-step: ALL B fragments loaded first (sched_barrier), then WMMAs
//     issue back-to-back -> one s_wait_dscnt per k-step instead of 16
//   - SpMM via global_atomic_add_f32 resolving in the 192MB L2
// ---------------------------------------------------------------------------

typedef __attribute__((ext_vector_type(16))) __bf16 v16bf;
typedef __attribute__((ext_vector_type(8)))  float  v8f;

#define N_NODES 50000
#define N_EDGES 800000
#define NFEAT   512
#define NHID    128
#define NCLASS  40
#define H_DIM   256   // 2*NHID
#define N1COLS  256   // fused output cols of GEMM1 (W1 | Wr1)
#define N2PAD   96    // fused padded cols of GEMM2 (48 for W2 | 48 for Wr2)
#define NT1     16    // n-tiles in GEMM1
#define NT2     6     // n-tiles in GEMM2
#define KSTEPS1 (NFEAT / 32)             // 16
#define ROWTILES ((N_NODES + 15) / 16)   // 3125 exactly

// ---- helpers ---------------------------------------------------------------

__device__ __forceinline__ unsigned short f2bf(float f) {
  union { float f; unsigned u; } x; x.f = f;
  unsigned r = x.u + 0x7FFFu + ((x.u >> 16) & 1u);  // round-to-nearest-even
  return (unsigned short)(r >> 16);
}
__device__ __forceinline__ unsigned pk2(float lo, float hi) {
  return (unsigned)f2bf(lo) | ((unsigned)f2bf(hi) << 16);
}

union BFrag { v16bf v; unsigned u[8]; };

// Hard scheduling fence: keep the load batch above, the WMMA batch below.
__device__ __forceinline__ void sched_fence() {
#if __has_builtin(__builtin_amdgcn_sched_barrier)
  __builtin_amdgcn_sched_barrier(0);
#else
  asm volatile("" ::: "memory");
#endif
}

// ---- weight prep: f32 [K][N] -> bf16 [N][K] (transposed, fused, padded) ----

__global__ void prep_wb1(const float* __restrict__ W1, const float* __restrict__ Wr1,
                         unsigned short* __restrict__ WB1) {
  int idx = blockIdx.x * blockDim.x + threadIdx.x;   // over 256*512
  if (idx >= N1COLS * NFEAT) return;
  int n = idx / NFEAT, k = idx - n * NFEAT;
  float v = (n < NHID) ? W1[(size_t)k * NHID + n] : Wr1[(size_t)k * NHID + (n - NHID)];
  WB1[idx] = f2bf(v);
}

__global__ void prep_wb2(const float* __restrict__ W2, const float* __restrict__ Wr2,
                         unsigned short* __restrict__ WB2) {
  int idx = blockIdx.x * blockDim.x + threadIdx.x;   // over 96*256
  if (idx >= N2PAD * H_DIM) return;
  int n = idx / H_DIM, k = idx - n * H_DIM;
  float v = 0.0f;
  if (n < NCLASS)                      v = W2 [(size_t)k * NCLASS + n];
  else if (n >= 48 && n < 48 + NCLASS) v = Wr2[(size_t)k * NCLASS + (n - 48)];
  WB2[idx] = f2bf(v);
}

__global__ void zero_f32(float* __restrict__ p, int n) {
  int i = blockIdx.x * blockDim.x + threadIdx.x;
  if (i < n) p[i] = 0.0f;
}

// ---- GEMM1: sup1 = x@W1 ; xr1 = x@Wr1 + br1 -------------------------------
// Workgroup = 8 waves = 128 rows; B k-slice (256n x 32k bf16 = 16KB) staged in
// LDS via async-to-LDS copies, double buffered, all 8 waves share it.

__global__ __launch_bounds__(256, 1) void gemm1_wmma(
    const float* __restrict__ x, const unsigned short* __restrict__ WB1,
    const float* __restrict__ br1,
    float* __restrict__ sup1, float* __restrict__ xr1) {
  __shared__ unsigned short Bsh[2][N1COLS * 32];     // 2 x 16KB

  const int tid  = threadIdx.x;
  const int wave = tid >> 5;
  const int lane = tid & 31;
  const int rowTile = blockIdx.x * 8 + wave;
  const bool valid = (rowTile < ROWTILES);           // wave-uniform
  const int row0 = valid ? rowTile * 16 : 0;
  const int half = lane >> 4;
  const int r    = lane & 15;

  v8f acc[NT1];
  #pragma unroll
  for (int i = 0; i < NT1; ++i)
    #pragma unroll
    for (int j = 0; j < 8; ++j) acc[i][j] = 0.0f;

  const float* arow = x + (size_t)(row0 + r) * NFEAT;

  // Cooperative async fill of one k-slice of B into Bsh[buf].
  // 256n x 64B = 1024 chunks of 16B; 256 threads -> 4 chunks each.
  auto fill_async = [&](int buf, int k0) {
    unsigned lbase = (unsigned)(uintptr_t)(&Bsh[buf][0]);
    #pragma unroll
    for (int q = 0; q < 4; ++q) {
      int chunk = q * 256 + tid;                  // 0..1023
      int n = chunk >> 2;
      int part = chunk & 3;
      const unsigned short* g = WB1 + (size_t)n * NFEAT + k0 + part * 8;
      unsigned l = lbase + (unsigned)(n * 64 + part * 16);
      asm volatile("global_load_async_to_lds_b128 %0, %1, off"
                   :: "v"(l), "v"(g) : "memory");
    }
  };

  fill_async(0, 0);

  for (int ks = 0; ks < KSTEPS1; ++ks) {
    const int cur = ks & 1;
    const int k0 = ks * 32;
    if (ks + 1 < KSTEPS1) {
      fill_async(1 - cur, k0 + 32);               // overlap next slice with compute
      asm volatile("s_wait_asynccnt 0x4" ::: "memory");  // current slice done
    } else {
      asm volatile("s_wait_asynccnt 0x0" ::: "memory");
    }
    __syncthreads();                              // slice visible to all waves

    if (valid) {
      __builtin_prefetch(arow + k0 + 32, 0, 0);   // global_prefetch_b8, next A slice
      // A fragment (16x32 bf16): lanes 0-15 -> K {0..7,16..23}; lanes 16-31 -> +8
      union { v16bf v; unsigned u[8]; } A;
      {
        const float4* p = (const float4*)(arow + k0 + half * 8);
        float4 c0 = p[0], c1 = p[1];
        const float4* q = (const float4*)(arow + k0 + 16 + half * 8);
        float4 c2 = q[0], c3 = q[1];
        A.u[0] = pk2(c0.x, c0.y); A.u[1] = pk2(c0.z, c0.w);
        A.u[2] = pk2(c1.x, c1.y); A.u[3] = pk2(c1.z, c1.w);
        A.u[4] = pk2(c2.x, c2.y); A.u[5] = pk2(c2.z, c2.w);
        A.u[6] = pk2(c3.x, c3.y); A.u[7] = pk2(c3.z, c3.w);
      }
      // Load ALL 16 B fragments (32 ds_load_b128, one dscnt wait), fence,
      // then stream 16 WMMAs back-to-back through the matrix pipe.
      const unsigned short* bbase = &Bsh[cur][half * 8];
      BFrag BB[NT1];
      #pragma unroll
      for (int nt = 0; nt < NT1; ++nt) {
        const unsigned short* bp = bbase + (nt * 16 + r) * 32;
        uint4 b0 = *(const uint4*)(bp);
        uint4 b1 = *(const uint4*)(bp + 16);
        BB[nt].u[0] = b0.x; BB[nt].u[1] = b0.y; BB[nt].u[2] = b0.z; BB[nt].u[3] = b0.w;
        BB[nt].u[4] = b1.x; BB[nt].u[5] = b1.y; BB[nt].u[6] = b1.z; BB[nt].u[7] = b1.w;
      }
      sched_fence();
      #pragma unroll
      for (int nt = 0; nt < NT1; ++nt)
        acc[nt] = __builtin_amdgcn_wmma_f32_16x16x32_bf16(
            false, A.v, false, BB[nt].v, (short)0, acc[nt], false, false);
    }
    __syncthreads();                              // reads done before next overwrite
  }

  if (valid) {
    #pragma unroll
    for (int nt = 0; nt < NT1; ++nt) {
      const int col = nt * 16 + r;                // 0..255
      #pragma unroll
      for (int v = 0; v < 8; ++v) {
        const int row = row0 + v + half * 8;
        float val = acc[nt][v];
        if (col < NHID) sup1[(size_t)row * NHID + col] = val;
        else            xr1 [(size_t)row * NHID + (col - NHID)] = val + br1[col - NHID];
      }
    }
  }
}

// ---- SpMM: out[dst] += val * sup[src]  (one wave per edge, atomics in L2) --

__global__ void spmm_atomic(const int* __restrict__ src, const int* __restrict__ dst,
                            const float* __restrict__ val, const float* __restrict__ sup,
                            float* __restrict__ out, int nE, int F) {
  int t = blockIdx.x * blockDim.x + threadIdx.x;
  int e = t >> 5;
  int lane = t & 31;
  if (e >= nE) return;
  int s = src[e], d = dst[e];
  float w = val[e];
  for (int f = lane; f < F; f += 32)
    atomicAdd(out + (size_t)d * F + f, w * sup[(size_t)s * F + f]);
}

// ---- h = concat(relu(gc1 + b1), xr1) stored as bf16 [N_NODES][256] --------

__global__ void build_h(const float* __restrict__ gc1, const float* __restrict__ b1,
                        const float* __restrict__ xr1, unsigned short* __restrict__ h) {
  int t = blockIdx.x * blockDim.x + threadIdx.x;     // over N_NODES*NHID
  if (t >= N_NODES * NHID) return;
  int node = t >> 7, j = t & (NHID - 1);
  float g = gc1[t] + b1[j];
  g = g > 0.0f ? g : 0.0f;
  h[(size_t)node * H_DIM + j]        = f2bf(g);
  h[(size_t)node * H_DIM + NHID + j] = f2bf(xr1[t]);
}

// ---- GEMM2: sup2 = h@W2 ; hr2 = h@Wr2 + br2  (A already bf16) -------------
// B (48KB) is fully L2-resident; direct global-load B fragments, batched
// ahead of the WMMAs so loads overlap the matrix pipe.

__global__ __launch_bounds__(256, 1) void gemm2_wmma(
    const unsigned short* __restrict__ h, const unsigned short* __restrict__ WB2,
    const float* __restrict__ br2,
    float* __restrict__ sup2, float* __restrict__ hr2) {
  const int wave = threadIdx.x >> 5;
  const int lane = threadIdx.x & 31;
  const int rowTile = blockIdx.x * 8 + wave;
  if (rowTile >= ROWTILES) return;            // wave-uniform: EXEC stays all-1s
  const int row0 = rowTile * 16;
  const int half = lane >> 4;
  const int r    = lane & 15;

  v8f acc[NT2];
  #pragma unroll
  for (int i = 0; i < NT2; ++i)
    #pragma unroll
    for (int j = 0; j < 8; ++j) acc[i][j] = 0.0f;

  const unsigned short* arow = h + (size_t)(row0 + r) * H_DIM;

  for (int k0 = 0; k0 < H_DIM; k0 += 32) {
    union { v16bf v; unsigned u[8]; } A;
    {
      uint4 q0 = *(const uint4*)(arow + k0 + half * 8);
      uint4 q1 = *(const uint4*)(arow + k0 + 16 + half * 8);
      A.u[0] = q0.x; A.u[1] = q0.y; A.u[2] = q0.z; A.u[3] = q0.w;
      A.u[4] = q1.x; A.u[5] = q1.y; A.u[6] = q1.z; A.u[7] = q1.w;
    }
    const unsigned short* bbase = WB2 + (size_t)r * H_DIM + k0 + half * 8;
    BFrag BB[NT2];
    #pragma unroll
    for (int nt = 0; nt < NT2; ++nt) {
      const unsigned short* bp = bbase + (size_t)nt * 16 * H_DIM;
      uint4 b0 = *(const uint4*)(bp);
      uint4 b1 = *(const uint4*)(bp + 16);
      BB[nt].u[0] = b0.x; BB[nt].u[1] = b0.y; BB[nt].u[2] = b0.z; BB[nt].u[3] = b0.w;
      BB[nt].u[4] = b1.x; BB[nt].u[5] = b1.y; BB[nt].u[6] = b1.z; BB[nt].u[7] = b1.w;
    }
    sched_fence();
    #pragma unroll
    for (int nt = 0; nt < NT2; ++nt)
      acc[nt] = __builtin_amdgcn_wmma_f32_16x16x32_bf16(
          false, A.v, false, BB[nt].v, (short)0, acc[nt], false, false);
  }

  #pragma unroll
  for (int nt = 0; nt < NT2; ++nt) {
    const int col = nt * 16 + r;              // 0..95
    #pragma unroll
    for (int v = 0; v < 8; ++v) {
      const int row = row0 + v + half * 8;
      float val = acc[nt][v];
      if (nt < 3) {
        if (col < NCLASS) sup2[(size_t)row * NCLASS + col] = val;
      } else {
        int c = col - 48;
        if (c < NCLASS) hr2[(size_t)row * NCLASS + c] = val + br2[c];
      }
    }
  }
}

// ---- final: log_softmax(gc2 + b2 + hr2) -----------------------------------

__global__ void final_logsoftmax(const float* __restrict__ gc2, const float* __restrict__ b2,
                                 const float* __restrict__ hr2, float* __restrict__ out) {
  int n = blockIdx.x * blockDim.x + threadIdx.x;
  if (n >= N_NODES) return;
  float v[NCLASS];
  float mx = -3.4e38f;
  #pragma unroll
  for (int c = 0; c < NCLASS; ++c) {
    float t = gc2[(size_t)n * NCLASS + c] + b2[c] + hr2[(size_t)n * NCLASS + c];
    v[c] = t;
    mx = fmaxf(mx, t);
  }
  float s = 0.0f;
  #pragma unroll
  for (int c = 0; c < NCLASS; ++c) s += __expf(v[c] - mx);
  float l = __logf(s);
  #pragma unroll
  for (int c = 0; c < NCLASS; ++c) out[(size_t)n * NCLASS + c] = v[c] - mx - l;
}

// ---------------------------------------------------------------------------

extern "C" void kernel_launch(void* const* d_in, const int* in_sizes, int n_in,
                              void* d_out, int out_size, void* d_ws, size_t ws_size,
                              hipStream_t stream) {
  const float* x        = (const float*)d_in[0];
  const int*   edge_src = (const int*)  d_in[1];
  const int*   edge_dst = (const int*)  d_in[2];
  const float* edge_val = (const float*)d_in[3];
  const float* W1  = (const float*)d_in[4];
  const float* b1  = (const float*)d_in[5];
  const float* Wr1 = (const float*)d_in[6];
  const float* br1 = (const float*)d_in[7];
  const float* W2  = (const float*)d_in[8];
  const float* b2  = (const float*)d_in[9];
  const float* Wr2 = (const float*)d_in[10];
  const float* br2 = (const float*)d_in[11];
  float* out = (float*)d_out;

  // Workspace layout (all offsets 256B aligned)
  char* ws = (char*)d_ws;
  unsigned short* WB1 = (unsigned short*)(ws);                       // 256*512 bf16 = 256 KB
  unsigned short* WB2 = (unsigned short*)(ws + 262144);              //  96*256 bf16 =  48 KB
  float* sup1 = (float*)(ws + 311296);                               // [N,128] f32
  float* xr1  = (float*)(ws + 311296 + 25600000);                    // [N,128] f32
  float* gc1  = (float*)(ws + 311296 + 2ull * 25600000);             // [N,128] f32 (accum)
  unsigned short* hbf = (unsigned short*)(ws + 311296 + 3ull * 25600000); // [N,256] bf16
  float* sup2 = (float*)(ws + 311296 + 4ull * 25600000);             // [N,40] f32
  float* hr2  = (float*)(ws + 311296 + 4ull * 25600000 + 8000000);   // [N,40] f32
  float* gc2  = (float*)(ws + 311296 + 4ull * 25600000 + 16000000);  // [N,40] f32 (accum)

  const int T = 256;

  // 0) weight prep (bf16, transposed [n][k], fused + padded)
  prep_wb1<<<(N1COLS * NFEAT + T - 1) / T, T, 0, stream>>>(W1, Wr1, WB1);
  prep_wb2<<<(N2PAD * H_DIM + T - 1) / T, T, 0, stream>>>(W2, Wr2, WB2);

  // zero accumulation buffers (fresh every call)
  zero_f32<<<(N_NODES * NHID + T - 1) / T, T, 0, stream>>>(gc1, N_NODES * NHID);
  zero_f32<<<(N_NODES * NCLASS + T - 1) / T, T, 0, stream>>>(gc2, N_NODES * NCLASS);

  // 1) fused GEMM: sup1 = x@W1, xr1 = x@Wr1 + br1   (async-LDS staged B)
  gemm1_wmma<<<(ROWTILES + 7) / 8, T, 0, stream>>>(x, WB1, br1, sup1, xr1);

  // 2) SpMM: gc1 += A @ sup1
  spmm_atomic<<<(N_EDGES * 32 + T - 1) / T, T, 0, stream>>>(
      edge_src, edge_dst, edge_val, sup1, gc1, N_EDGES, NHID);

  // 3) h = concat(relu(gc1 + b1), xr1) as bf16
  build_h<<<(N_NODES * NHID + T - 1) / T, T, 0, stream>>>(gc1, b1, xr1, hbf);

  // 4) fused GEMM: sup2 = h@W2, hr2 = h@Wr2 + br2
  gemm2_wmma<<<(ROWTILES + 7) / 8, T, 0, stream>>>(hbf, WB2, br2, sup2, hr2);

  // 5) SpMM: gc2 += A @ sup2
  spmm_atomic<<<(N_EDGES * 32 + T - 1) / T, T, 0, stream>>>(
      edge_src, edge_dst, edge_val, sup2, gc2, N_EDGES, NCLASS);

  // 6) out = log_softmax(gc2 + b2 + hr2)
  final_logsoftmax<<<(N_NODES + T - 1) / T, T, 0, stream>>>(gc2, b2, hr2, out);
}
